// NonLocalBlock2D_with_mask_Res_27290222198898
// MI455X (gfx1250) — compile-verified, matrix operand order checked by reading, not run on hardware
//
#include <hip/hip_runtime.h>

typedef _Float16 half_t;
typedef __attribute__((ext_vector_type(16))) _Float16 v16h;
typedef __attribute__((ext_vector_type(8)))  float    v8f;

#define NB   4
#define CH   64
#define HW   64
#define NPIX 4096   // 64*64

// Wave-relative LDS byte offset of a __shared__ object (addrspace(3) offset).
__device__ __forceinline__ unsigned lds_off(void* p) {
    return (unsigned)(uintptr_t)(__attribute__((address_space(3))) void*)p;
}

// ---------------------------------------------------------------------------
// 1) Fused 1x1 projections -> f16 theta[B,N,64], phiT[B,N,64], maskedV[B,N,64]
//    mask_i simplifies to (1 - mask); fold it into V = mask_i * g.
// ---------------------------------------------------------------------------
__global__ __launch_bounds__(64) void proj_kernel(
    const float* __restrict__ x, const float* __restrict__ mask,
    const float* __restrict__ gw, const float* __restrict__ gb,
    const float* __restrict__ tw, const float* __restrict__ tb,
    const float* __restrict__ pw, const float* __restrict__ pb,
    half_t* __restrict__ thetaH, half_t* __restrict__ phiH,
    half_t* __restrict__ gmH)
{
    const int bn = blockIdx.x;            // b*NPIX + n
    const int b  = bn >> 12;
    const int n  = bn & 4095;
    const int t  = threadIdx.x;           // out channel
    __shared__ float xs[64];
    xs[t] = x[(size_t)(b * 64 + t) * NPIX + n];
    __syncthreads();
    float aT = tb[t], aP = pb[t], aG = gb[t];
#pragma unroll 8
    for (int c = 0; c < 64; ++c) {
        const float xv = xs[c];
        aT += tw[t * 64 + c] * xv;
        aP += pw[t * 64 + c] * xv;
        aG += gw[t * 64 + c] * xv;
    }
    const float mi = 1.0f - mask[b * NPIX + n];    // mask_i
    const size_t o = (size_t)bn * 64 + t;
    thetaH[o] = (half_t)aT;
    phiH[o]   = (half_t)aP;
    gmH[o]    = (half_t)(aG * mi);
}

// ---------------------------------------------------------------------------
// 2) Flash attention:  y[n,:] = (sum_m exp(q.k_m) * Vmask[m,:]) / (sum_m exp)
//    Online softmax; WMMA f32_16x16x32_f16 for both QK^T and PV.
//    K/V chunks double-buffered in LDS via GLOBAL_LOAD_ASYNC_TO_LDS_B128
//    (ASYNCcnt), overlapping the next chunk's fetch with WMMA + softmax.
//    Block: 256 threads = 8 waves, each wave owns 16 query rows (QBLOCK=128).
// ---------------------------------------------------------------------------
__global__ __launch_bounds__(256) void attn_kernel(
    const half_t* __restrict__ thetaH, const half_t* __restrict__ phiH,
    const half_t* __restrict__ gmH, float* __restrict__ y)
{
    const int b    = blockIdx.y;
    const int wave = threadIdx.x >> 5;
    const int lane = threadIdx.x & 31;
    const int q0   = blockIdx.x * 128 + wave * 16;

    __shared__ half_t Ks[2][32][72];     // double-buffered 32 keys x 64 dims
    __shared__ half_t Vs[2][32][72];     // (+skew; 144B row stride, 16B aligned)
    __shared__ float  sc[8][16][33];     // per-wave 16x32 score tile (+pad)
    __shared__ float  rs[8][16];         // per-wave row scale / rowsum

    const int hsel = lane >> 4;          // lane half select
    const int l15  = lane & 15;
    const int kb   = hsel * 8;           // ISA A/B fragment K base

    // ---- Q fragments (16x64 split into two 16x32 K-chunks) ----
    const half_t* qrow = thetaH + ((size_t)(b * NPIX) + q0 + l15) * 64;
    v16h aQ0, aQ1;
#pragma unroll
    for (int j = 0; j < 8; ++j) {
        aQ0[j]     = qrow[kb + j];
        aQ0[8 + j] = qrow[16 + kb + j];
        aQ1[j]     = qrow[32 + kb + j];
        aQ1[8 + j] = qrow[48 + kb + j];
    }

    // ---- async copy of one 32x64 K/V chunk (1 x B128 per thread per mat) ----
    const int tkey = threadIdx.x >> 3;       // 0..31
    const int td0  = (threadIdx.x & 7) * 8;  // 0..56
    const half_t* kbase = phiH + (size_t)(b * NPIX) * 64;
    const half_t* vbase = gmH  + (size_t)(b * NPIX) * 64;
    auto issue_chunk = [&](int kc, int buf) {
        const half_t* ksrc = kbase + (size_t)(kc * 32 + tkey) * 64 + td0;
        const half_t* vsrc = vbase + (size_t)(kc * 32 + tkey) * 64 + td0;
        unsigned kdst = lds_off(&Ks[buf][tkey][td0]);
        unsigned vdst = lds_off(&Vs[buf][tkey][td0]);
        asm volatile("global_load_async_to_lds_b128 %0, %1, off"
                     :: "v"(kdst), "v"(ksrc) : "memory");
        asm volatile("global_load_async_to_lds_b128 %0, %1, off"
                     :: "v"(vdst), "v"(vsrc) : "memory");
    };

    v8f oacc[4] = {v8f{}, v8f{}, v8f{}, v8f{}};   // 16x64 output accumulator
    float m_run = -1e30f, l_run = 0.0f;
    const int r = lane >> 1, hh = lane & 1;       // 2 lanes per softmax row

    issue_chunk(0, 0);                            // prologue

    for (int kc = 0; kc < 128; ++kc) {
        const int cur = kc & 1;
        // chunk kc resident in Ks/Vs[cur]; all waves done with buffer cur^1
        asm volatile("s_wait_asynccnt 0x0" ::: "memory");
        __syncthreads();
        if (kc + 1 < 128) issue_chunk(kc + 1, cur ^ 1);   // overlap next fetch

        // ---- S = Q @ K^T for two 16-key subtiles ----
#pragma unroll
        for (int sub = 0; sub < 2; ++sub) {
            const int keyc = sub * 16 + l15;        // B fragment col = key
            v16h bK0, bK1;
#pragma unroll
            for (int j = 0; j < 8; ++j) {
                bK0[j]     = Ks[cur][keyc][kb + j];
                bK0[8 + j] = Ks[cur][keyc][16 + kb + j];
                bK1[j]     = Ks[cur][keyc][32 + kb + j];
                bK1[8 + j] = Ks[cur][keyc][48 + kb + j];
            }
            v8f s = {};
            s = __builtin_amdgcn_wmma_f32_16x16x32_f16(false, aQ0, false, bK0,
                                                       (short)0, s, false, false);
            s = __builtin_amdgcn_wmma_f32_16x16x32_f16(false, aQ1, false, bK1,
                                                       (short)0, s, false, false);
#pragma unroll
            for (int i = 0; i < 8; ++i)
                sc[wave][i + 8 * hsel][sub * 16 + l15] = s[i];
        }

        // ---- online softmax (wave-local; LDS is in-order per wave) ----
        float mloc = -1e30f;
#pragma unroll
        for (int j = 0; j < 16; ++j) mloc = fmaxf(mloc, sc[wave][r][hh * 16 + j]);
        mloc = fmaxf(mloc, __shfl_xor(mloc, 1, 32));
        const float mnew  = fmaxf(m_run, mloc);
        const float scale = __expf(m_run - mnew);
        float psum = 0.0f;
#pragma unroll
        for (int j = 0; j < 16; ++j) {
            const float e = __expf(sc[wave][r][hh * 16 + j] - mnew);
            sc[wave][r][hh * 16 + j] = e;
            psum += e;
        }
        psum += __shfl_xor(psum, 1, 32);
        l_run = l_run * scale + psum;     // UNMASKED denominator (mask is in V)
        m_run = mnew;
        rs[wave][r] = scale;

        // ---- rescale O rows by exp(m_old - m_new) ----
#pragma unroll
        for (int i = 0; i < 8; ++i) {
            const float scl = rs[wave][i + 8 * hsel];
            oacc[0][i] *= scl; oacc[1][i] *= scl;
            oacc[2][i] *= scl; oacc[3][i] *= scl;
        }

        // ---- P fragment (16 queries x 32 keys, f16) ----
        v16h aP;
#pragma unroll
        for (int j = 0; j < 8; ++j) {
            aP[j]     = (half_t)sc[wave][l15][kb + j];
            aP[8 + j] = (half_t)sc[wave][l15][16 + kb + j];
        }
        // ---- O += P @ Vmask  (4 column tiles of 16) ----
#pragma unroll
        for (int ct = 0; ct < 4; ++ct) {
            const int col = ct * 16 + l15;
            v16h bV;
#pragma unroll
            for (int j = 0; j < 8; ++j) {
                bV[j]     = Vs[cur][kb + j][col];
                bV[8 + j] = Vs[cur][16 + kb + j][col];
            }
            oacc[ct] = __builtin_amdgcn_wmma_f32_16x16x32_f16(false, aP, false, bV,
                                                              (short)0, oacc[ct],
                                                              false, false);
        }
    }

    // ---- finalize: divide by unmasked softmax denominator, store y[B,N,64] ----
    rs[wave][r] = l_run;
#pragma unroll
    for (int i = 0; i < 8; ++i) {
        const int rowl = i + 8 * hsel;
        const float inv = 1.0f / rs[wave][rowl];
        const size_t base = ((size_t)(b * NPIX) + q0 + rowl) * 64 + l15;
        y[base]      = oacc[0][i] * inv;
        y[base + 16] = oacc[1][i] * inv;
        y[base + 32] = oacc[2][i] * inv;
        y[base + 48] = oacc[3][i] * inv;
    }
}

// ---------------------------------------------------------------------------
// 3) W_y = conv1x1(y) -> NCHW f32
// ---------------------------------------------------------------------------
__global__ __launch_bounds__(64) void wy_kernel(
    const float* __restrict__ y, const float* __restrict__ ww,
    const float* __restrict__ wb, float* __restrict__ out)
{
    const int bn = blockIdx.x;
    const int b  = bn >> 12;
    const int n  = bn & 4095;
    const int t  = threadIdx.x;
    __shared__ float ys[64];
    ys[t] = y[(size_t)bn * 64 + t];
    __syncthreads();
    float a = wb[t];
#pragma unroll 8
    for (int ic = 0; ic < 64; ++ic) a += ww[t * 64 + ic] * ys[ic];
    out[(size_t)(b * 64 + t) * NPIX + n] = a;
}

// ---------------------------------------------------------------------------
// 4) 3x3 conv with reflect-pad(1), LDS-tiled 66x66 input plane per (b,oc).
// ---------------------------------------------------------------------------
__device__ __forceinline__ int refl(int v) { return v < 0 ? -v : (v > 63 ? 126 - v : v); }

__global__ __launch_bounds__(256) void conv3x3_kernel(
    const float* __restrict__ in, const float* __restrict__ w,
    const float* __restrict__ bias, float* __restrict__ out)
{
    const int bc = blockIdx.x;
    const int b = bc >> 6, oc = bc & 63;
    const int t = threadIdx.x;
    __shared__ float tile[66 * 66];
    float acc[16];
    const float bv = bias[oc];
#pragma unroll
    for (int i = 0; i < 16; ++i) acc[i] = bv;

    for (int ic = 0; ic < 64; ++ic) {
        const float* src = in + (size_t)(b * 64 + ic) * NPIX;
        for (int idx = t; idx < 66 * 66; idx += 256) {
            const int py = idx / 66, px = idx % 66;
            tile[idx] = src[refl(py - 1) * 64 + refl(px - 1)];
        }
        __syncthreads();
        const float* wp = w + (size_t)(oc * 64 + ic) * 9;   // uniform -> s_loads
        const float w0 = wp[0], w1 = wp[1], w2 = wp[2];
        const float w3 = wp[3], w4 = wp[4], w5 = wp[5];
        const float w6 = wp[6], w7 = wp[7], w8 = wp[8];
#pragma unroll
        for (int i = 0; i < 16; ++i) {
            const int pid = i * 256 + t;
            const int py = pid >> 6, px = pid & 63;
            const float* tp = &tile[py * 66 + px];
            acc[i] += w0 * tp[0]   + w1 * tp[1]   + w2 * tp[2]
                    + w3 * tp[66]  + w4 * tp[67]  + w5 * tp[68]
                    + w6 * tp[132] + w7 * tp[133] + w8 * tp[134];
        }
        __syncthreads();
    }
    float* dst = out + (size_t)(b * 64 + oc) * NPIX;
#pragma unroll
    for (int i = 0; i < 16; ++i) dst[i * 256 + t] = acc[i];
}

// ---------------------------------------------------------------------------
// 5) InstanceNorm stats (mean, rsqrt(var+eps)) per (b,c) plane.
// ---------------------------------------------------------------------------
__global__ __launch_bounds__(256) void in_stats_kernel(
    const float* __restrict__ v, float* __restrict__ stats)
{
    const int bc = blockIdx.x;
    const float* p = v + (size_t)bc * NPIX;
    float s = 0.0f, s2 = 0.0f;
    for (int i = threadIdx.x; i < NPIX; i += 256) {
        const float x = p[i];
        s += x; s2 += x * x;
    }
#pragma unroll
    for (int off = 16; off > 0; off >>= 1) {
        s  += __shfl_xor(s,  off, 32);
        s2 += __shfl_xor(s2, off, 32);
    }
    __shared__ float ls[8], ls2[8];
    const int wv = threadIdx.x >> 5, ln = threadIdx.x & 31;
    if (ln == 0) { ls[wv] = s; ls2[wv] = s2; }
    __syncthreads();
    if (threadIdx.x == 0) {
        float ts = 0.0f, ts2 = 0.0f;
        for (int i = 0; i < 8; ++i) { ts += ls[i]; ts2 += ls2[i]; }
        const float mu  = ts / (float)NPIX;
        const float var = ts2 / (float)NPIX - mu * mu;
        stats[bc * 2]     = mu;
        stats[bc * 2 + 1] = rsqrtf(var + 1e-5f);
    }
}

// mode 0: relu(norm(v));  mode 1: res + norm(v)
__global__ __launch_bounds__(256) void in_apply_kernel(
    const float* __restrict__ v, const float* __restrict__ stats,
    const float* __restrict__ res, float* __restrict__ out, int mode)
{
    const int i  = blockIdx.x * 256 + threadIdx.x;
    const int bc = i >> 12;
    const float h = (v[i] - stats[bc * 2]) * stats[bc * 2 + 1];
    out[i] = (mode == 0) ? fmaxf(h, 0.0f) : res[i] + h;
}

// ---------------------------------------------------------------------------
// 6) z = mask_i * x + (1 - mask_i) * W_y  with mask_i = 1 - mask
// ---------------------------------------------------------------------------
__global__ __launch_bounds__(256) void blend_kernel(
    const float* __restrict__ x, const float* __restrict__ mask,
    const float* __restrict__ wy, float* __restrict__ z)
{
    const int i = blockIdx.x * 256 + threadIdx.x;   // over B*C*N
    const int b = i >> 18;                          // 64*4096 per batch
    const int n = i & 4095;
    const float mi = 1.0f - mask[b * NPIX + n];
    z[i] = mi * x[i] + (1.0f - mi) * wy[i];
}

// ---------------------------------------------------------------------------
extern "C" void kernel_launch(void* const* d_in, const int* in_sizes, int n_in,
                              void* d_out, int out_size, void* d_ws, size_t ws_size,
                              hipStream_t stream) {
    (void)in_sizes; (void)n_in; (void)out_size; (void)ws_size;
    const float* x    = (const float*)d_in[0];
    const float* mask = (const float*)d_in[1];
    const float* g_w  = (const float*)d_in[2];
    const float* g_b  = (const float*)d_in[3];
    const float* t_w  = (const float*)d_in[4];
    const float* t_b  = (const float*)d_in[5];
    const float* p_w  = (const float*)d_in[6];
    const float* p_b  = (const float*)d_in[7];
    const float* w_w  = (const float*)d_in[8];
    const float* w_b  = (const float*)d_in[9];
    const float* rw1  = (const float*)d_in[10];
    const float* rb1  = (const float*)d_in[11];
    const float* rw2  = (const float*)d_in[12];
    const float* rb2  = (const float*)d_in[13];

    char* ws = (char*)d_ws;
    half_t* thetaH = (half_t*)(ws + (0ull  << 20));  // 2 MB
    half_t* phiH   = (half_t*)(ws + (2ull  << 20));  // 2 MB
    half_t* gmH    = (half_t*)(ws + (4ull  << 20));  // 2 MB
    float*  yb     = (float*) (ws + (6ull  << 20));  // 4 MB
    float*  bufA   = (float*) (ws + (10ull << 20));  // 4 MB
    float*  bufB   = (float*) (ws + (14ull << 20));  // 4 MB
    float*  bufC   = (float*) (ws + (18ull << 20));  // 4 MB
    float*  stats  = (float*) (ws + (22ull << 20));  // 2 KB

    proj_kernel<<<NB * NPIX, 64, 0, stream>>>(x, mask, g_w, g_b, t_w, t_b,
                                              p_w, p_b, thetaH, phiH, gmH);
    attn_kernel<<<dim3(NPIX / 128, NB), 256, 0, stream>>>(thetaH, phiH, gmH, yb);
    wy_kernel<<<NB * NPIX, 64, 0, stream>>>(yb, w_w, w_b, bufA);

    for (int i = 0; i < 3; ++i) {
        conv3x3_kernel<<<NB * 64, 256, 0, stream>>>(bufA, rw1 + (size_t)i * 64 * 64 * 9,
                                                    rb1 + i * 64, bufB);
        in_stats_kernel<<<NB * 64, 256, 0, stream>>>(bufB, stats);
        in_apply_kernel<<<4096, 256, 0, stream>>>(bufB, stats, nullptr, bufC, 0);
        conv3x3_kernel<<<NB * 64, 256, 0, stream>>>(bufC, rw2 + (size_t)i * 64 * 64 * 9,
                                                    rb2 + i * 64, bufB);
        in_stats_kernel<<<NB * 64, 256, 0, stream>>>(bufB, stats);
        in_apply_kernel<<<4096, 256, 0, stream>>>(bufB, stats, bufA, bufA, 1);
    }
    blend_kernel<<<4096, 256, 0, stream>>>(x, mask, bufA, (float*)d_out);
}
